// Transformer_84748294685171
// MI455X (gfx1250) — matrix operand order repaired and checked by reading, not compiled, not run on hardware
//
#include <hip/hip_runtime.h>
#include <hip/hip_bf16.h>
#include <math.h>

// ---------------------------------------------------------------------------
// CDNA5 / gfx1250 transformer forward.
//  * All GEMMs: one batched WMMA kernel, v_wmma_f32_16x16x32_f16, f32 accum.
//  * All GEMM operands pre-converted to f16; B stored [N][K] (k contiguous).
//  * LDS staging via GLOBAL_LOAD_ASYNC_TO_LDS_B128 (ASYNCcnt), double-buffered:
//    prefetch tile k+1 while computing tile k; wait asynccnt<=3 to consume.
//  * Block tile 128x64, BK=32; each of 8 waves owns 32x32 -> 4 WMMAs/K-step.
// ---------------------------------------------------------------------------

typedef __attribute__((ext_vector_type(16))) _Float16 v16h;
typedef __attribute__((ext_vector_type(8)))  _Float16 v8h;
typedef __attribute__((ext_vector_type(8)))  float    v8f;

#define BM 128
#define BN 64
#define BK 32
#define KPAD 40   // halves; row stride 80B (multiple of 16B)

// C output modes
#define CM_F32   0   // float,  C[m*ldc + n]
#define CM_F16   1   // f16,    C[m*ldc + n]
#define CM_F16T  2   // f16,    C[n*ldc + m]   (transposed store)

static __device__ __forceinline__ void async_b128(unsigned lds_addr, const void* gaddr)
{
    asm volatile("global_load_async_to_lds_b128 %0, %1, off"
                 :: "v"(lds_addr), "v"(gaddr) : "memory");
}
static __device__ __forceinline__ void async_b128_off16(unsigned lds_addr, const void* gaddr)
{
    asm volatile("global_load_async_to_lds_b128 %0, %1, off offset:16"
                 :: "v"(lds_addr), "v"(gaddr) : "memory");
}
template <int N>
static __device__ __forceinline__ void wait_async()
{
#if __has_builtin(__builtin_amdgcn_s_wait_asynccnt)
    __builtin_amdgcn_s_wait_asynccnt(N);
#else
    asm volatile("s_wait_asynccnt %0" :: "i"(N) : "memory");
#endif
}

// ---------------------------------------------------------------------------
// Batched GEMM: C[batch] = A(MxK,f16) * B([N][K],f16) (+bias f32)(+relu)
// Batch z = (outer = z/Hin, inner = z%Hin) with separate strides.
// M%128==0, N%64==0, K%32==0.  A: k contiguous (lda).  B: k contiguous (ldb).
// ---------------------------------------------------------------------------
__global__ __launch_bounds__(256)
void gemm_wmma_f16(const _Float16* __restrict__ A, int lda, long long sAo, long long sAi,
                   const _Float16* __restrict__ Bm, int ldb, long long sBo, long long sBi,
                   void* __restrict__ Cv, int ldc, long long sCo, long long sCi,
                   const float* __restrict__ bias, int K, int Hin, int relu, int cmode)
{
    __shared__ alignas(16) _Float16 As[2][BM][KPAD];   // 2 x 10240 B
    __shared__ alignas(16) _Float16 Bs[2][BN][KPAD];   // 2 x  5120 B

    const int tid = threadIdx.x;
    const int bz  = blockIdx.z;
    const int bo  = bz / Hin, bi = bz % Hin;
    const long long offA = (long long)bo * sAo + (long long)bi * sAi;
    const long long offB = (long long)bo * sBo + (long long)bi * sBi;
    const long long offC = (long long)bo * sCo + (long long)bi * sCi;
    const int m0 = blockIdx.y * BM;
    const int n0 = blockIdx.x * BN;

    const int lane = tid & 31;
    const int wv   = tid >> 5;
    const int wm   = wv & 3;        // 4 x 32 = 128 rows
    const int wn   = wv >> 2;       // 2 x 32 = 64 cols
    const int lr   = lane & 15;
    const int g    = lane >> 4;

    // staging maps (pure f16 byte copies)
    const int ar = tid >> 1;            // A row 0..127
    const int ak = (tid & 1) * 16;      // A k-halves segment {0,16} (32B each)
    const int bn = tid >> 2;            // B row (n) 0..63
    const int bk = (tid & 3) * 8;       // B k-halves segment (16B each)

    const _Float16* gA0 = A  + offA + (long long)(m0 + ar) * lda + ak;
    const _Float16* gB0 = Bm + offB + (long long)(n0 + bn) * ldb + bk;
    const unsigned ldsA[2] = { (unsigned)(size_t)&As[0][ar][ak],
                               (unsigned)(size_t)&As[1][ar][ak] };
    const unsigned ldsB[2] = { (unsigned)(size_t)&Bs[0][bn][bk],
                               (unsigned)(size_t)&Bs[1][bn][bk] };

    auto issue_tile = [&](int kk, int buf) {
        async_b128(ldsA[buf], gA0 + kk);          // 16B: halves ak..ak+7
        async_b128_off16(ldsA[buf], gA0 + kk);    // 16B: halves ak+8..ak+15
        async_b128(ldsB[buf], gB0 + kk);          // 16B
    };

    v8f acc[2][2] = {};

    issue_tile(0, 0);                              // prologue prefetch
    int cur = 0;

    for (int k0 = 0; k0 < K; k0 += BK) {
        if (k0 + BK < K) {
            issue_tile(k0 + BK, cur ^ 1);          // prefetch next tile
            wait_async<3>();                       // current tile's 3 copies done
        } else {
            wait_async<0>();
        }
        __syncthreads();                           // all waves' copies visible

        // ---- fragments per ISA 16-bit layouts (two aligned v8h loads each)
        union { v16h v; v8h h[2]; } a[2], b[2];
        #pragma unroll
        for (int i = 0; i < 2; ++i) {
            const int row = wm * 32 + i * 16 + lr;
            a[i].h[0] = *(const v8h*)&As[cur][row][g * 8];
            a[i].h[1] = *(const v8h*)&As[cur][row][16 + g * 8];
        }
        #pragma unroll
        for (int j = 0; j < 2; ++j) {
            const int colr = wn * 32 + j * 16 + lr;
            b[j].h[0] = *(const v8h*)&Bs[cur][colr][g * 8];
            b[j].h[1] = *(const v8h*)&Bs[cur][colr][16 + g * 8];
        }

        #pragma unroll
        for (int i = 0; i < 2; ++i)
            #pragma unroll
            for (int j = 0; j < 2; ++j)
                acc[i][j] = __builtin_amdgcn_wmma_f32_16x16x32_f16(
                    false, a[i].v, false, b[j].v, (short)0, acc[i][j], false, false);

        __syncthreads();   // fragment reads done before this buf is re-filled
        cur ^= 1;
    }

    #pragma unroll
    for (int j = 0; j < 2; ++j) {
        const int col = n0 + wn * 32 + j * 16 + lr;
        const float bv = bias ? bias[col] : 0.0f;
        #pragma unroll
        for (int i = 0; i < 2; ++i) {
            #pragma unroll
            for (int r = 0; r < 8; ++r) {
                const int mr = m0 + wm * 32 + i * 16 + g * 8 + r;  // m = r + 8g
                float v = acc[i][j][r] + bv;
                if (relu) v = v > 0.0f ? v : 0.0f;
                if (cmode == CM_F32)
                    ((float*)Cv)[offC + (long long)mr * ldc + col] = v;
                else if (cmode == CM_F16)
                    ((_Float16*)Cv)[offC + (long long)mr * ldc + col] = (_Float16)v;
                else
                    ((_Float16*)Cv)[offC + (long long)col * ldc + mr] = (_Float16)v;
            }
        }
    }
}

// ---------------------------------------------------------------------------
// Weight convert + transpose:  in f32 [K][N]  ->  out f16 [N][K]
// K%32==0, N%32==0.  Block 256 = 32x8, 32x32 tile via LDS.
// ---------------------------------------------------------------------------
__global__ __launch_bounds__(256)
void convert_transpose(const float* __restrict__ in, _Float16* __restrict__ out,
                       int K, int N)
{
    __shared__ float tile[32][33];
    const int nt = blockIdx.x * 32;
    const int kt = blockIdx.y * 32;
    const int tx = threadIdx.x & 31;
    const int ty = threadIdx.x >> 5;   // 0..7
    #pragma unroll
    for (int i = 0; i < 32; i += 8)
        tile[ty + i][tx] = in[(long long)(kt + ty + i) * N + (nt + tx)];
    __syncthreads();
    #pragma unroll
    for (int i = 0; i < 32; i += 8)
        out[(long long)(nt + ty + i) * K + (kt + tx)] = (_Float16)tile[tx][ty + i];
}

// ---------------------------------------------------------------------------
// Masked, scaled softmax: reads f32 scores, writes f16 probabilities.
// mode 0: none | 1: mask[b][k] (B,1,1,Sk) | 2: mask[b][q][k] (B,1,Sq,Sk)
// ---------------------------------------------------------------------------
__global__ __launch_bounds__(256)
void softmax_rows(const float* __restrict__ in, _Float16* __restrict__ outh,
                  const float* __restrict__ mask, int mode,
                  int Hh, int Sq, int Sk, float scale)
{
    __shared__ float red[256];
    const int row = blockIdx.x;              // bh*Sq + q
    const int bh  = row / Sq;
    const int qi  = row - bh * Sq;
    const int b   = bh / Hh;
    const float* p = in + (long long)row * Sk;
    const float* mr = nullptr;
    if (mode == 1)      mr = mask + (long long)b * Sk;
    else if (mode == 2) mr = mask + ((long long)b * Sq + qi) * Sk;

    const int t = threadIdx.x;
    float loc[4];
    int cnt = 0;
    float mx = -3.4e38f;
    for (int i = t; i < Sk; i += 256) {
        float v = p[i] * scale;
        if (mr) v += mr[i] * -1e9f;
        loc[cnt++] = v;
        mx = fmaxf(mx, v);
    }
    red[t] = mx; __syncthreads();
    for (int o = 128; o > 0; o >>= 1) {
        if (t < o) red[t] = fmaxf(red[t], red[t + o]);
        __syncthreads();
    }
    mx = red[0]; __syncthreads();

    float sm = 0.0f;
    cnt = 0;
    for (int i = t; i < Sk; i += 256) {
        float e = expf(loc[cnt] - mx);
        loc[cnt++] = e;
        sm += e;
    }
    red[t] = sm; __syncthreads();
    for (int o = 128; o > 0; o >>= 1) {
        if (t < o) red[t] += red[t + o];
        __syncthreads();
    }
    const float inv = 1.0f / red[0];
    cnt = 0;
    for (int i = t; i < Sk; i += 256)
        outh[(long long)row * Sk + i] = (_Float16)(loc[cnt++] * inv);
}

// ---------------------------------------------------------------------------
// out = LayerNorm(z + x) * gamma + beta (eps=1e-3); writes f32 + f16 copies.
// Safe in-place on x (values cached in registers before the barrier).
// ---------------------------------------------------------------------------
__global__ __launch_bounds__(256)
void ln_residual(const float* __restrict__ z, const float* __restrict__ x,
                 const float* __restrict__ gg, const float* __restrict__ bb,
                 float* __restrict__ out, _Float16* __restrict__ outh, int W)
{
    __shared__ float r1[256], r2[256];
    const int row = blockIdx.x, t = threadIdx.x;
    const float* zp = z + (long long)row * W;
    const float* xp = x + (long long)row * W;
    float loc[4];
    int cnt = 0;
    float s = 0.0f, s2 = 0.0f;
    for (int i = t; i < W; i += 256) {
        float v = zp[i] + xp[i];
        loc[cnt++] = v;
        s += v; s2 += v * v;
    }
    r1[t] = s; r2[t] = s2; __syncthreads();
    for (int o = 128; o > 0; o >>= 1) {
        if (t < o) { r1[t] += r1[t + o]; r2[t] += r2[t + o]; }
        __syncthreads();
    }
    const float mean = r1[0] / W;
    const float var  = r2[0] / W - mean * mean;
    const float inv  = rsqrtf(var + 1e-3f);
    cnt = 0;
    for (int i = t; i < W; i += 256) {
        float v = gg[i] * (loc[cnt++] - mean) * inv + bb[i];
        out[(long long)row * W + i]  = v;
        outh[(long long)row * W + i] = (_Float16)v;
    }
}

// ---------------------------------------------------------------------------
// out[row] = emb[ids[row]] * sqrt(E) + PE[row % Slen]; f32 + f16 copies.
// ---------------------------------------------------------------------------
__global__ __launch_bounds__(256)
void embed_pe(const int* __restrict__ ids, const float* __restrict__ emb,
              float* __restrict__ out, _Float16* __restrict__ outh,
              int Slen, int Ew, float scale)
{
    const int row = blockIdx.x;
    const int s   = row % Slen;
    const int id  = ids[row];
    const int t   = threadIdx.x;
    for (int e = t; e < Ew; e += 256) {
        float di  = (float)(2 * (e >> 1)) / (float)Ew;
        float ang = (float)s * powf(10000.0f, -di);
        float pe  = (e & 1) ? cosf(ang) : sinf(ang);
        float v   = emb[(long long)id * Ew + e] * scale + pe;
        out[(long long)row * Ew + e]  = v;
        outh[(long long)row * Ew + e] = (_Float16)v;
    }
}

// ===========================================================================
// Host-side orchestration
// ===========================================================================
namespace {

constexpr int Bsz = 8, Ssz = 512, Tsz = 512, Esz = 512, Hsz = 8, DHsz = 64;
constexpr int Fsz = 2048, Vtar = 32000, Vsrc = 32000;
constexpr int NLAY = 6;

static inline void gemm_launch(hipStream_t st,
    const _Float16* A, int lda, long long sAo, long long sAi,
    const _Float16* Bm, int ldb, long long sBo, long long sBi,
    void* C, int ldc, long long sCo, long long sCi,
    const float* bias, int M, int N, int K, int batches, int Hin,
    int relu, int cmode)
{
    dim3 grid(N / BN, M / BM, batches);
    gemm_wmma_f16<<<grid, 256, 0, st>>>(A, lda, sAo, sAi, Bm, ldb, sBo, sBi,
                                        C, ldc, sCo, sCi, bias, K, Hin,
                                        relu, cmode);
}

// Parameter leaves, JAX sorted-dict-key order:
//   0 W_out, 1 b_out, 2+26l dec layers, 158 emb_src, 159 emb_tar,
//   160+16l enc layers.
struct Params {
    void* const* din;
    int n_in;
    long long off[256];

    void build_offsets() {
        long long sz[256];
        int i = 0;
        auto mha_sz = [&]() {
            sz[i++] = (long long)Esz * Esz; sz[i++] = (long long)Esz * Esz;
            sz[i++] = (long long)Esz * Esz; sz[i++] = (long long)Esz * Esz;
            sz[i++] = Esz; sz[i++] = Esz; sz[i++] = Esz; sz[i++] = Esz;
        };
        sz[i++] = (long long)Esz * Vtar;
        sz[i++] = Vtar;
        for (int l = 0; l < NLAY; ++l) {
            sz[i++] = (long long)Esz * Fsz; sz[i++] = (long long)Fsz * Esz;
            sz[i++] = Fsz; sz[i++] = Esz;
            for (int j = 0; j < 6; ++j) sz[i++] = Esz;
            mha_sz(); mha_sz();
        }
        sz[i++] = (long long)Vsrc * Esz;
        sz[i++] = (long long)Vtar * Esz;
        for (int l = 0; l < NLAY; ++l) {
            sz[i++] = (long long)Esz * Fsz; sz[i++] = (long long)Fsz * Esz;
            sz[i++] = Fsz; sz[i++] = Esz;
            for (int j = 0; j < 4; ++j) sz[i++] = Esz;
            mha_sz();
        }
        long long a = 0;
        for (int k = 0; k < 256; ++k) { off[k] = a; a += sz[k]; }
    }

    const float* leaf(int i) const {
        if (n_in > 6) return (const float*)din[5 + i];
        return (const float*)din[5] + off[i];
    }
};

struct MhaW {  // f16 [N][K] weights + f32 biases
    const _Float16 *WQ, *WK, *WV, *WO;
    const float *bQ, *bK, *bV, *bO;
};

// scratch f16 buffers shared by all attention calls
struct MhaBufs { _Float16 *qh, *kh, *vth, *ch, *sch; float* sc; };

static void run_mha(hipStream_t st, const _Float16* xqh, const _Float16* xkvh,
                    int Sq, int Skv, const MhaW& w,
                    const float* mask, int mmode, const MhaBufs& t, float* outz)
{
    // Q,K: (B*S,E)x(E,E) -> f16
    gemm_launch(st, xqh,  Esz, 0, 0, w.WQ, Esz, 0, 0, t.qh, Esz, 0, 0, w.bQ,
                Bsz * Sq,  Esz, Esz, 1, 1, 0, CM_F16);
    gemm_launch(st, xkvh, Esz, 0, 0, w.WK, Esz, 0, 0, t.kh, Esz, 0, 0, w.bK,
                Bsz * Skv, Esz, Esz, 1, 1, 0, CM_F16);
    // V: batched over b, stored transposed -> vth[b][E][Skv] (k contiguous)
    gemm_launch(st, xkvh, Esz, (long long)Skv * Esz, 0,
                w.WV, Esz, 0, 0,
                t.vth, Skv, (long long)Esz * Skv, 0, w.bV,
                Skv, Esz, Esz, Bsz, 1, 0, CM_F16T);

    // scores[b,h] = Q_head (Sq x DH) * K_head^T -> f32 sc
    gemm_launch(st,
                t.qh, Esz, (long long)Sq  * Esz, DHsz,
                t.kh, Esz, (long long)Skv * Esz, DHsz,
                t.sc, Skv, (long long)Hsz * Sq * Skv, (long long)Sq * Skv,
                nullptr, Sq, Skv, DHsz, Bsz * Hsz, Hsz, 0, CM_F32);

    softmax_rows<<<Bsz * Hsz * Sq, 256, 0, st>>>(t.sc, t.sch, mask, mmode,
                                                 Hsz, Sq, Skv, 0.125f);

    // context[b,h] = attn (Sq x Skv) * V_head; heads interleaved into (B,Sq,E)
    gemm_launch(st,
                t.sch, Skv, (long long)Hsz * Sq * Skv, (long long)Sq * Skv,
                t.vth, Skv, (long long)Esz * Skv, (long long)DHsz * Skv,
                t.ch,  Esz, (long long)Sq * Esz, DHsz,
                nullptr, Sq, DHsz, Skv, Bsz * Hsz, Hsz, 0, CM_F16);

    // output projection -> f32 zb
    gemm_launch(st, t.ch, Esz, 0, 0, w.WO, Esz, 0, 0, outz, Esz, 0, 0, w.bO,
                Bsz * Sq, Esz, Esz, 1, 1, 0, CM_F32);
}

} // namespace

extern "C" void kernel_launch(void* const* d_in, const int* in_sizes, int n_in,
                              void* d_out, int out_size, void* d_ws, size_t ws_size,
                              hipStream_t stream)
{
    (void)in_sizes; (void)out_size; (void)ws_size;

    const int*   inp_enc = (const int*)d_in[0];
    const int*   inp_dec = (const int*)d_in[1];
    const float* enc_pad = (const float*)d_in[2];
    const float* la_mask = (const float*)d_in[3];
    const float* dec_pad = (const float*)d_in[4];

    Params P{d_in, n_in, {}};
    P.build_offsets();

    const int DEC0 = 2;
    const int EMB_SRC = 2 + 26 * NLAY;   // 158
    const int EMB_TAR = EMB_SRC + 1;     // 159
    const int ENC0 = EMB_TAR + 1;        // 160

    // ---- workspace carve ------------------------------------------------
    char* wsp = (char*)d_ws;
    auto allocF = [&](size_t n) { float* p = (float*)wsp;    wsp += n * 4; return p; };
    auto allocH = [&](size_t n) { _Float16* p = (_Float16*)wsp; wsp += n * 2; return p; };

    const size_t NX = (size_t)Bsz * Ssz * Esz;            // 2M elements
    float*    enc_x  = allocF(NX);
    float*    dec_x  = allocF(NX);
    float*    zb     = allocF(NX);
    float*    sc     = allocF((size_t)Bsz * Hsz * Ssz * Ssz);
    _Float16* enc_xh = allocH(NX);
    _Float16* dec_xh = allocH(NX);
    _Float16* qh     = allocH(NX);
    _Float16* kh     = allocH(NX);
    _Float16* vth    = allocH(NX);
    _Float16* ch     = allocH(NX);
    _Float16* ffnh   = allocH((size_t)Bsz * Ssz * Fsz);
    _Float16* sch    = allocH((size_t)Bsz * Hsz * Ssz * Ssz);

    // ---- weight conversion: f32 [K][N] -> f16 [N][K] --------------------
    auto conv = [&](const float* w, int K, int N) -> const _Float16* {
        _Float16* o = allocH((size_t)K * N);
        dim3 grid(N / 32, K / 32);
        convert_transpose<<<grid, 256, 0, stream>>>(w, o, K, N);
        return o;
    };

    const _Float16* Woutt = conv(P.leaf(0), Esz, Vtar);
    const float*    b_out = P.leaf(1);

    MhaW encW[NLAY]; const _Float16 *encW1[NLAY], *encW2[NLAY];
    MhaW decW1[NLAY], decW2[NLAY]; const _Float16 *decF1[NLAY], *decF2[NLAY];

    for (int l = 0; l < NLAY; ++l) {
        const int b = ENC0 + 16 * l;
        encW1[l] = conv(P.leaf(b + 0), Esz, Fsz);
        encW2[l] = conv(P.leaf(b + 1), Fsz, Esz);
        encW[l]  = { conv(P.leaf(b + 10), Esz, Esz), conv(P.leaf(b + 8),  Esz, Esz),
                     conv(P.leaf(b + 11), Esz, Esz), conv(P.leaf(b + 9),  Esz, Esz),
                     P.leaf(b + 14), P.leaf(b + 12), P.leaf(b + 15), P.leaf(b + 13) };
    }
    for (int l = 0; l < NLAY; ++l) {
        const int b = DEC0 + 26 * l;
        decF1[l] = conv(P.leaf(b + 0), Esz, Fsz);
        decF2[l] = conv(P.leaf(b + 1), Fsz, Esz);
        decW1[l] = { conv(P.leaf(b + 12), Esz, Esz), conv(P.leaf(b + 10), Esz, Esz),
                     conv(P.leaf(b + 13), Esz, Esz), conv(P.leaf(b + 11), Esz, Esz),
                     P.leaf(b + 16), P.leaf(b + 14), P.leaf(b + 17), P.leaf(b + 15) };
        decW2[l] = { conv(P.leaf(b + 20), Esz, Esz), conv(P.leaf(b + 18), Esz, Esz),
                     conv(P.leaf(b + 21), Esz, Esz), conv(P.leaf(b + 19), Esz, Esz),
                     P.leaf(b + 24), P.leaf(b + 22), P.leaf(b + 25), P.leaf(b + 23) };
    }

    MhaBufs T{qh, kh, vth, ch, sch, sc};

    const float escale = sqrtf((float)Esz);
    embed_pe<<<Bsz * Ssz, 256, 0, stream>>>(inp_enc, P.leaf(EMB_SRC),
                                            enc_x, enc_xh, Ssz, Esz, escale);
    embed_pe<<<Bsz * Tsz, 256, 0, stream>>>(inp_dec, P.leaf(EMB_TAR),
                                            dec_x, dec_xh, Tsz, Esz, escale);

    const int RowsS = Bsz * Ssz;

    // ---- encoder --------------------------------------------------------
    for (int l = 0; l < NLAY; ++l) {
        const int b = ENC0 + 16 * l;
        run_mha(stream, enc_xh, enc_xh, Ssz, Ssz, encW[l], enc_pad, 1, T, zb);
        ln_residual<<<RowsS, 256, 0, stream>>>(zb, enc_x, P.leaf(b + 5),
                                               P.leaf(b + 4), enc_x, enc_xh, Esz);
        gemm_launch(stream, enc_xh, Esz, 0, 0, encW1[l], Esz, 0, 0,
                    ffnh, Fsz, 0, 0, P.leaf(b + 2), RowsS, Fsz, Esz, 1, 1, 1, CM_F16);
        gemm_launch(stream, ffnh, Fsz, 0, 0, encW2[l], Fsz, 0, 0,
                    zb, Esz, 0, 0, P.leaf(b + 3), RowsS, Esz, Fsz, 1, 1, 0, CM_F32);
        ln_residual<<<RowsS, 256, 0, stream>>>(zb, enc_x, P.leaf(b + 7),
                                               P.leaf(b + 6), enc_x, enc_xh, Esz);
    }

    // ---- decoder --------------------------------------------------------
    for (int l = 0; l < NLAY; ++l) {
        const int b = DEC0 + 26 * l;
        run_mha(stream, dec_xh, dec_xh, Tsz, Tsz, decW1[l], la_mask, 2, T, zb);
        ln_residual<<<RowsS, 256, 0, stream>>>(zb, dec_x, P.leaf(b + 5),
                                               P.leaf(b + 4), dec_x, dec_xh, Esz);

        run_mha(stream, dec_xh, enc_xh, Tsz, Ssz, decW2[l], dec_pad, 1, T, zb);
        ln_residual<<<RowsS, 256, 0, stream>>>(zb, dec_x, P.leaf(b + 7),
                                               P.leaf(b + 6), dec_x, dec_xh, Esz);

        gemm_launch(stream, dec_xh, Esz, 0, 0, decF1[l], Esz, 0, 0,
                    ffnh, Fsz, 0, 0, P.leaf(b + 2), RowsS, Fsz, Esz, 1, 1, 1, CM_F16);
        gemm_launch(stream, ffnh, Fsz, 0, 0, decF2[l], Fsz, 0, 0,
                    zb, Esz, 0, 0, P.leaf(b + 3), RowsS, Esz, Fsz, 1, 1, 0, CM_F32);
        ln_residual<<<RowsS, 256, 0, stream>>>(zb, dec_x, P.leaf(b + 9),
                                               P.leaf(b + 8), dec_x, dec_xh, Esz);
    }

    // ---- final vocab projection -> d_out (B*T, V_TAR) f32 ---------------
    gemm_launch(stream, dec_xh, Esz, 0, 0, Woutt, Esz, 0, 0,
                (float*)d_out, Vtar, 0, 0, b_out,
                Bsz * Tsz, Vtar, Esz, 1, 1, 0, CM_F32);
}
